// Decoder_35442070127330
// MI455X (gfx1250) — matrix operand order repaired and checked by reading, not compiled, not run on hardware
//
#include <hip/hip_runtime.h>

// ---------------------------------------------------------------------------
// CDNA5 (gfx1250) GRU decoder: bf16 WMMA 16x16x32, fused GRU step kernel.
// ---------------------------------------------------------------------------

typedef __bf16 bf16_t;
typedef __attribute__((ext_vector_type(8)))  __bf16 bf16x8;
typedef __attribute__((ext_vector_type(16))) __bf16 bf16x16;
typedef __attribute__((ext_vector_type(8)))  float  f32x8;

#define DEV static __device__ __forceinline__

// ---- model dims -----------------------------------------------------------
static constexpr int BATCH = 512;
static constexpr int TSTEP = 120;
static constexpr int LAT   = 292;
static constexpr int H1    = 701, HP1 = 704;   // pad to mult of 32
static constexpr int H3    = 901, HP3 = 928;
static constexpr int NCHAR = 35;

// ---- bf16 helpers (bit-exact RNE, no reliance on hw cvt availability) -----
DEV bf16_t f2bf(float f) {
  unsigned u = __builtin_bit_cast(unsigned, f);
  u += 0x7FFFu + ((u >> 16) & 1u);
  unsigned short s = (unsigned short)(u >> 16);
  return __builtin_bit_cast(bf16_t, s);
}
DEV float bf2f(bf16_t b) {
  unsigned short s = __builtin_bit_cast(unsigned short, b);
  return __builtin_bit_cast(float, (unsigned)s << 16);
}

// A fragment (16x32 bf16): lane holds K {kb..kb+7} and {kb+16..kb+23}
DEV bf16x16 load_a_frag(const bf16_t* p) {
  bf16x8 lo = *(const bf16x8*)p;
  bf16x8 hi = *(const bf16x8*)(p + 16);
  bf16x16 r;
#pragma unroll
  for (int i = 0; i < 8; ++i) { r[i] = lo[i]; r[i + 8] = hi[i]; }
  return r;
}
// B fragment (32x16 bf16): lane = N column, K {kb..kb+7} and {kb+16..kb+23} strided by ld
DEV bf16x16 load_b_frag(const bf16_t* p, int ld) {
  bf16x16 r;
#pragma unroll
  for (int i = 0; i < 8; ++i) { r[i] = p[(size_t)i * ld]; r[i + 8] = p[(size_t)(16 + i) * ld]; }
  return r;
}
DEV f32x8 wmma_bf16(bf16x16 a, bf16x16 b, f32x8 c) {
  return __builtin_amdgcn_wmma_f32_16x16x32_bf16(false, a, false, b,
                                                 (short)0, c, false, false);
}
DEV float sigmoidf_(float x) { return 1.0f / (1.0f + __expf(-x)); }

// ---------------------------------------------------------------------------
// util: zero fill (dwords)
// ---------------------------------------------------------------------------
__global__ void k_zero(unsigned* __restrict__ p, long n) {
  long i = (long)blockIdx.x * blockDim.x + threadIdx.x;
  if (i < n) p[i] = 0u;
}

// ---------------------------------------------------------------------------
// dense_1 + SELU: x[b,d] = selu(z[b,:]·W1[:,d] + b1[d])   (43 MFLOP, scalar)
// ---------------------------------------------------------------------------
__global__ void k_dense1(const float* __restrict__ z, const float* __restrict__ W1,
                         const float* __restrict__ b1, float* __restrict__ x) {
  int i = blockIdx.x * blockDim.x + threadIdx.x;
  if (i >= BATCH * LAT) return;
  int b = i / LAT, d = i % LAT;
  float acc = b1[d];
  const float* zr = z + (size_t)b * LAT;
  for (int k = 0; k < LAT; ++k) acc += zr[k] * W1[(size_t)k * LAT + d];
  const float scale = 1.0507009873554805f, alpha = 1.6732632423543772f;
  x[i] = scale * (acc > 0.f ? acc : alpha * (__expf(acc) - 1.f));
}

// ---------------------------------------------------------------------------
// xp1[b,c] = x[b,:]·gWi1[:,c] + gbi1[c]  (time-invariant: RepeatVector input)
// ---------------------------------------------------------------------------
__global__ void k_xp1(const float* __restrict__ x, const float* __restrict__ Wi,
                      const float* __restrict__ bi, float* __restrict__ xp) {
  int i = blockIdx.x * blockDim.x + threadIdx.x;
  if (i >= BATCH * 3 * H1) return;
  int b = i / (3 * H1), c = i % (3 * H1);
  float acc = bi[c];
  const float* xr = x + (size_t)b * LAT;
  for (int k = 0; k < LAT; ++k) acc += xr[k] * Wi[(size_t)k * (3 * H1) + c];
  xp[i] = acc;
}

// ---------------------------------------------------------------------------
// weight staging: W fp32 [K][3H] -> bf16 [Kpad][3*Hpad] gate-split, padded
// ---------------------------------------------------------------------------
__global__ void k_convw(const float* __restrict__ W, bf16_t* __restrict__ Wc,
                        int K, int H, int Kpad, int Hpad) {
  int total = Kpad * 3 * Hpad;
  int i = blockIdx.x * blockDim.x + threadIdx.x;
  if (i >= total) return;
  int k = i / (3 * Hpad);
  int r = i % (3 * Hpad);
  int g = r / Hpad, n = r % Hpad;
  float v = (k < K && n < H) ? W[(size_t)k * (3 * H) + g * H + n] : 0.f;
  Wc[i] = f2bf(v);
}

// ---------------------------------------------------------------------------
// Fused GRU step (reset_after, gates z,r,h):
//   x-GEMM (optional) + h-GEMM with bf16 WMMA, gate math in epilogue.
// Block: 256 thr (8 waves). WG tile 64x32; wave tile 16x16 per accumulator.
// grid = (Hpad/32, BATCH/64)
// ---------------------------------------------------------------------------
template <bool XGEMM>
__global__ __launch_bounds__(256) void k_gru_step(
    const bf16_t* __restrict__ xA, int xRowStride, int Kx,
    const bf16_t* __restrict__ WiC,       // [Kx][3*Hpad] bf16
    const float* __restrict__ xp,         // [B][3*H] fp32 (used if !XGEMM)
    const bf16_t* __restrict__ hA, int Kh,// h_{t-1} bf16 [B][Hpad] (Kh==Hpad)
    const bf16_t* __restrict__ WhC,       // [Hpad][3*Hpad] bf16
    const float* __restrict__ bi, const float* __restrict__ bh,
    const float* __restrict__ hprev, float* __restrict__ hnext, // fp32 [B][H]
    bf16_t* __restrict__ hnextb,          // bf16 [B][Hpad]
    bf16_t* __restrict__ seqOut, int outRowStride,
    int H, int Hpad) {
  const int tid  = threadIdx.x;
  const int wave = tid >> 5, lane = tid & 31;
  const int msub = wave & 3, nsub = wave >> 2;
  const int m0 = blockIdx.y * 64 + msub * 16;
  const int n0 = blockIdx.x * 32 + nsub * 16;
  const int laneLo  = lane & 15;
  const int laneHiK = (lane >> 4) << 3;  // 0 or 8
  const int wst = 3 * Hpad;

  f32x8 axz = {0,0,0,0,0,0,0,0}, axr = {0,0,0,0,0,0,0,0}, axh = {0,0,0,0,0,0,0,0};
  f32x8 ahz = {0,0,0,0,0,0,0,0}, ahr = {0,0,0,0,0,0,0,0}, ahh = {0,0,0,0,0,0,0,0};

  // ---- recurrent GEMM: h_{t-1} @ Wh ----
  {
    const bf16_t* aRow = hA + (size_t)(m0 + laneLo) * Kh + laneHiK;
    for (int kc = 0; kc < Kh; kc += 32) {
      bf16x16 af = load_a_frag(aRow + kc);
      const bf16_t* bRow = WhC + (size_t)(kc + laneHiK) * wst + n0 + laneLo;
      ahz = wmma_bf16(af, load_b_frag(bRow, wst),            ahz);
      ahr = wmma_bf16(af, load_b_frag(bRow + Hpad, wst),     ahr);
      ahh = wmma_bf16(af, load_b_frag(bRow + 2 * Hpad, wst), ahh);
    }
  }
  // ---- input GEMM: x_t @ Wi ----
  if (XGEMM) {
    const bf16_t* aRow = xA + (size_t)(m0 + laneLo) * xRowStride + laneHiK;
    for (int kc = 0; kc < Kx; kc += 32) {
      bf16x16 af = load_a_frag(aRow + kc);
      const bf16_t* bRow = WiC + (size_t)(kc + laneHiK) * wst + n0 + laneLo;
      axz = wmma_bf16(af, load_b_frag(bRow, wst),            axz);
      axr = wmma_bf16(af, load_b_frag(bRow + Hpad, wst),     axr);
      axh = wmma_bf16(af, load_b_frag(bRow + 2 * Hpad, wst), axh);
    }
  }

  // ---- epilogue: gate math + state update ----
  // C/D layout: VGPR r, lanes0-15 -> M=r, lanes16-31 -> M=8+r, N=lane&15
  const int mbase = m0 + ((lane >> 4) << 3);
  const int ncol  = n0 + laneLo;
#pragma unroll
  for (int r = 0; r < 8; ++r) {
    const int m = mbase + r;
    float hnv = 0.f;
    if (ncol < H) {
      float xz, xr, xh;
      if (XGEMM) {
        xz = axz[r] + bi[ncol];
        xr = axr[r] + bi[H + ncol];
        xh = axh[r] + bi[2 * H + ncol];
      } else {
        const float* xrow = xp + (size_t)m * (3 * H);
        xz = xrow[ncol]; xr = xrow[H + ncol]; xh = xrow[2 * H + ncol];
      }
      float rz = ahz[r] + bh[ncol];
      float rr = ahr[r] + bh[H + ncol];
      float rh = ahh[r] + bh[2 * H + ncol];
      float zg = sigmoidf_(xz + rz);
      float rg = sigmoidf_(xr + rr);
      float hh = tanhf(xh + rg * rh);
      float hp = hprev[(size_t)m * H + ncol];
      hnv = zg * hp + (1.f - zg) * hh;
      hnext[(size_t)m * H + ncol] = hnv;
    }
    bf16_t hb16 = f2bf(hnv);                       // pads write 0 -> keeps K clean
    hnextb[(size_t)m * Hpad + ncol] = hb16;
    seqOut[(size_t)m * outRowStride + ncol] = hb16;
  }
}

// ---------------------------------------------------------------------------
// output dense + softmax: one block per (b,t) row
// ---------------------------------------------------------------------------
__global__ __launch_bounds__(64) void k_out(const bf16_t* __restrict__ h3,
                                            const float* __restrict__ Wo,
                                            const float* __restrict__ bo,
                                            float* __restrict__ out) {
  const int row = blockIdx.x;              // b*T + t
  const bf16_t* hr = h3 + (size_t)row * HP3;
  __shared__ float lg[NCHAR];
  const int c = threadIdx.x;
  if (c < NCHAR) {
    float acc = bo[c];
    for (int k = 0; k < H3; ++k) acc += bf2f(hr[k]) * Wo[(size_t)k * NCHAR + c];
    lg[c] = acc;
  }
  __syncthreads();
  if (c < NCHAR) {
    float mx = lg[0];
    for (int j = 1; j < NCHAR; ++j) mx = fmaxf(mx, lg[j]);
    float s = 0.f;
    for (int j = 0; j < NCHAR; ++j) s += __expf(lg[j] - mx);
    out[(size_t)row * NCHAR + c] = __expf(lg[c] - mx) / s;
  }
}

// ---------------------------------------------------------------------------
extern "C" void kernel_launch(void* const* d_in, const int* in_sizes, int n_in,
                              void* d_out, int out_size, void* d_ws, size_t ws_size,
                              hipStream_t stream) {
  const float* z    = (const float*)d_in[0];
  const float* W1   = (const float*)d_in[1];
  const float* b1   = (const float*)d_in[2];
  const float* gWi1 = (const float*)d_in[3];
  const float* gWh1 = (const float*)d_in[4];
  const float* gbi1 = (const float*)d_in[5];
  const float* gbh1 = (const float*)d_in[6];
  const float* gWi2 = (const float*)d_in[7];
  const float* gWh2 = (const float*)d_in[8];
  const float* gbi2 = (const float*)d_in[9];
  const float* gbh2 = (const float*)d_in[10];
  const float* gWi3 = (const float*)d_in[11];
  const float* gWh3 = (const float*)d_in[12];
  const float* gbi3 = (const float*)d_in[13];
  const float* gbh3 = (const float*)d_in[14];
  const float* Wo   = (const float*)d_in[15];
  const float* bo   = (const float*)d_in[16];
  float* out = (float*)d_out;

  char* ws = (char*)d_ws;
  size_t off = 0;
  auto carve = [&](size_t bytes) -> char* {
    char* p = ws + off;
    off = (off + bytes + 255) & ~(size_t)255;
    return p;
  };

  float*  x_f32 = (float*)carve((size_t)BATCH * LAT * 4);
  float*  xp1   = (float*)carve((size_t)BATCH * 3 * H1 * 4);
  bf16_t* Wc_h1 = (bf16_t*)carve((size_t)HP1 * 3 * HP1 * 2);
  bf16_t* Wc_i2 = (bf16_t*)carve((size_t)HP1 * 3 * HP1 * 2);
  bf16_t* Wc_h2 = (bf16_t*)carve((size_t)HP1 * 3 * HP1 * 2);
  bf16_t* Wc_i3 = (bf16_t*)carve((size_t)HP1 * 3 * HP3 * 2);
  bf16_t* Wc_h3 = (bf16_t*)carve((size_t)HP3 * 3 * HP3 * 2);
  bf16_t* seq1  = (bf16_t*)carve((size_t)BATCH * TSTEP * HP1 * 2);
  bf16_t* seq2  = (bf16_t*)carve((size_t)BATCH * TSTEP * HP1 * 2);
  bf16_t* seq3  = (bf16_t*)carve((size_t)BATCH * TSTEP * HP3 * 2);
  float*  hf0   = (float*)carve((size_t)BATCH * HP3 * 4);
  float*  hf1   = (float*)carve((size_t)BATCH * HP3 * 4);
  bf16_t* hb0   = (bf16_t*)carve((size_t)BATCH * HP3 * 2);
  bf16_t* hb1   = (bf16_t*)carve((size_t)BATCH * HP3 * 2);
  float*  hf[2] = {hf0, hf1};
  bf16_t* hb[2] = {hb0, hb1};

  auto zero = [&](void* p, size_t bytes) {
    long n = (long)(bytes / 4);
    k_zero<<<dim3((unsigned)((n + 255) / 256)), 256, 0, stream>>>((unsigned*)p, n);
  };

  // --- stage activations + bf16 weights -----------------------------------
  {
    int n = BATCH * LAT;
    k_dense1<<<dim3((n + 255) / 256), 256, 0, stream>>>(z, W1, b1, x_f32);
  }
  {
    int n = BATCH * 3 * H1;
    k_xp1<<<dim3((n + 255) / 256), 256, 0, stream>>>(x_f32, gWi1, gbi1, xp1);
  }
  auto convw = [&](const float* W, bf16_t* Wc, int K, int H, int Kpad, int Hpad) {
    int n = Kpad * 3 * Hpad;
    k_convw<<<dim3((n + 255) / 256), 256, 0, stream>>>(W, Wc, K, H, Kpad, Hpad);
  };
  convw(gWh1, Wc_h1, H1, H1, HP1, HP1);
  convw(gWi2, Wc_i2, H1, H1, HP1, HP1);
  convw(gWh2, Wc_h2, H1, H1, HP1, HP1);
  convw(gWi3, Wc_i3, H1, H3, HP1, HP3);
  convw(gWh3, Wc_h3, H3, H3, HP3, HP3);

  // --- layer 1 (XGEMM=false: time-invariant xp1) ---------------------------
  zero(hf[0], (size_t)BATCH * H1 * 4);
  zero(hb[0], (size_t)BATCH * HP1 * 2);
  for (int t = 0; t < TSTEP; ++t) {
    int ri = t & 1, wi = (t + 1) & 1;
    k_gru_step<false><<<dim3(HP1 / 32, BATCH / 64), 256, 0, stream>>>(
        (const bf16_t*)nullptr, 0, 0, (const bf16_t*)nullptr, xp1,
        hb[ri], HP1, Wc_h1, gbi1, gbh1,
        hf[ri], hf[wi], hb[wi],
        seq1 + (size_t)t * HP1, TSTEP * HP1, H1, HP1);
  }
  // --- layer 2 -------------------------------------------------------------
  zero(hf[0], (size_t)BATCH * H1 * 4);
  zero(hb[0], (size_t)BATCH * HP1 * 2);
  for (int t = 0; t < TSTEP; ++t) {
    int ri = t & 1, wi = (t + 1) & 1;
    k_gru_step<true><<<dim3(HP1 / 32, BATCH / 64), 256, 0, stream>>>(
        seq1 + (size_t)t * HP1, TSTEP * HP1, HP1, Wc_i2, (const float*)nullptr,
        hb[ri], HP1, Wc_h2, gbi2, gbh2,
        hf[ri], hf[wi], hb[wi],
        seq2 + (size_t)t * HP1, TSTEP * HP1, H1, HP1);
  }
  // --- layer 3 -------------------------------------------------------------
  zero(hf[0], (size_t)BATCH * H3 * 4);
  zero(hb[0], (size_t)BATCH * HP3 * 2);
  for (int t = 0; t < TSTEP; ++t) {
    int ri = t & 1, wi = (t + 1) & 1;
    k_gru_step<true><<<dim3(HP3 / 32, BATCH / 64), 256, 0, stream>>>(
        seq2 + (size_t)t * HP1, TSTEP * HP1, HP1, Wc_i3, (const float*)nullptr,
        hb[ri], HP3, Wc_h3, gbi3, gbh3,
        hf[ri], hf[wi], hb[wi],
        seq3 + (size_t)t * HP3, TSTEP * HP3, H3, HP3);
  }
  // --- output dense + softmax ---------------------------------------------
  k_out<<<dim3(BATCH * TSTEP), 64, 0, stream>>>(seq3, Wo, bo, out);

  (void)in_sizes; (void)n_in; (void)out_size; (void)ws_size;
}
